// FactorizationMachineModel_65712999629184
// MI455X (gfx1250) — compile-verified, matrix operand order checked by reading, not run on hardware
//
#include <hip/hip_runtime.h>

// FM: out[b] = g_bias + sum_f mask*bias + 0.5*(||sum_f mask*emb||^2 - sum_f mask*||emb||^2)
// Implemented as mask[1024,160000] @ {emb16 | (rowsq,bias,0...)} via WMMA f16.

typedef _Float16 v16h __attribute__((ext_vector_type(16)));
typedef float    v8f  __attribute__((ext_vector_type(8)));

#define B_ROWS   1024
#define F_DIM    160000
#define KCHUNK   32
#define NCHUNKS  (F_DIM / KCHUNK)    // 5000
#define MTILES   (B_ROWS / 16)       // 64
#define SLICES   16
#define WAVES    8
#define KSTRIDE  (SLICES * WAVES)    // 128 chunk stride

// ISA 7.12.2 16-bit matrix per-lane K ordering (A 16x32 / B 32x16 symmetric):
// lane<16 : halves 0..7 -> K 0..7,  halves 8..15 -> K 16..23
// lane>=16: halves 0..7 -> K 8..15, halves 8..15 -> K 24..31
__device__ __forceinline__ int koff(int lane, int h) {
    int hg = h >> 3, hi = h & 7;
    return (lane < 16) ? (hg ? 16 + hi : hi) : (hg ? 24 + hi : 8 + hi);
}

// ---------------- pack B panels (f16, pre-swizzled to WMMA lane layout) ----------------
__global__ void fm_pack(const float* __restrict__ emb, const float* __restrict__ bias,
                        _Float16* __restrict__ embp, _Float16* __restrict__ auxp) {
    int id = blockIdx.x * blockDim.x + threadIdx.x;
    if (id >= NCHUNKS * 32) return;
    int ck   = id >> 5;
    int lane = id & 31;
    int n    = lane & 15;          // B-matrix column (N)
    v16h e, a;
#pragma unroll
    for (int h = 0; h < 16; ++h) {
        int k = ck * KCHUNK + koff(lane, h);
        e[h] = (_Float16)emb[(size_t)k * 16 + n];
        float av = 0.0f;
        if (n == 0) {
            float s = 0.0f;
#pragma unroll
            for (int d = 0; d < 16; ++d) { float v = emb[(size_t)k * 16 + d]; s += v * v; }
            av = s;                 // rowsq
        } else if (n == 1) {
            av = bias[k];           // linear bias
        }
        a[h] = (_Float16)av;
    }
    *(v16h*)(embp + (size_t)id * 16) = e;
    *(v16h*)(auxp + (size_t)id * 16) = a;
}

// ---------------- zero accumulator tiles ----------------
__global__ void fm_zero(float* __restrict__ c) {
    int i = blockIdx.x * blockDim.x + threadIdx.x;
    if (i < MTILES * 512) c[i] = 0.0f;
}

// ---------------- main GEMM: mask @ panels via v_wmma_f32_16x16x32_f16 ----------------
__device__ __forceinline__ void cvt8(const int4& a, const int4& b, v16h& d, int base) {
    d[base + 0] = a.x > 0 ? (_Float16)1.0f : (_Float16)0.0f;
    d[base + 1] = a.y > 0 ? (_Float16)1.0f : (_Float16)0.0f;
    d[base + 2] = a.z > 0 ? (_Float16)1.0f : (_Float16)0.0f;
    d[base + 3] = a.w > 0 ? (_Float16)1.0f : (_Float16)0.0f;
    d[base + 4] = b.x > 0 ? (_Float16)1.0f : (_Float16)0.0f;
    d[base + 5] = b.y > 0 ? (_Float16)1.0f : (_Float16)0.0f;
    d[base + 6] = b.z > 0 ? (_Float16)1.0f : (_Float16)0.0f;
    d[base + 7] = b.w > 0 ? (_Float16)1.0f : (_Float16)0.0f;
}

__global__ void __launch_bounds__(256) fm_main(const int* __restrict__ x,
                                               const _Float16* __restrict__ embp,
                                               const _Float16* __restrict__ auxp,
                                               float* __restrict__ ctile) {
    __shared__ float red[WAVES][512];

    const int tid   = threadIdx.x;
    const int lane  = tid & 31;
    const int wave  = tid >> 5;
    const int tileM = blockIdx.x & (MTILES - 1);
    const int slice = blockIdx.x >> 6;

    const int    row     = tileM * 16 + (lane & 15);     // A-matrix row for this lane
    const size_t rowBase = (size_t)row * F_DIM;
    const int    segA    = (lane < 16) ? 0 : 8;          // per-lane K sub-offset

    v8f acc0 = {};   // mask @ emb   -> ev tile
    v8f acc1 = {};   // mask @ aux   -> (sumsq, lin, 0...) tile

    for (int c = slice * WAVES + wave; c < NCHUNKS; c += KSTRIDE) {
        const int  k0 = c * KCHUNK;
        const int* px = x + rowBase + k0 + segA;
        int4 q0 = *(const int4*)(px);
        int4 q1 = *(const int4*)(px + 4);
        int4 q2 = *(const int4*)(px + 16);
        int4 q3 = *(const int4*)(px + 20);

        if (c + KSTRIDE < NCHUNKS)                        // uniform per wave
            __builtin_prefetch((const void*)(px + (size_t)KSTRIDE * KCHUNK), 0, 1);

        v16h a;
        cvt8(q0, q1, a, 0);   // K segment 1 (8 contiguous ints)
        cvt8(q2, q3, a, 8);   // K segment 2

        const v16h be = *(const v16h*)(embp + (size_t)c * 512 + lane * 16);
        const v16h ba = *(const v16h*)(auxp + (size_t)c * 512 + lane * 16);

        acc0 = __builtin_amdgcn_wmma_f32_16x16x32_f16(false, a, false, be, (short)0, acc0, false, false);
        acc1 = __builtin_amdgcn_wmma_f32_16x16x32_f16(false, a, false, ba, (short)0, acc1, false, false);
    }

    // C layout: lanes 0-15 -> M = v, N = lane; lanes 16-31 -> M = v+8, N = lane-16
    const int n   = lane & 15;
    const int mhi = (lane >> 4) * 8;
#pragma unroll
    for (int v = 0; v < 8; ++v) {
        int m = v + mhi;
        red[wave][m * 16 + n]       = acc0[v];
        red[wave][256 + m * 16 + n] = acc1[v];
    }
    __syncthreads();

    float s0 = 0.0f, s1 = 0.0f;
#pragma unroll
    for (int w = 0; w < WAVES; ++w) { s0 += red[w][tid]; s1 += red[w][256 + tid]; }
    atomicAdd(ctile + (size_t)tileM * 512 + tid,       s0);
    atomicAdd(ctile + (size_t)tileM * 512 + 256 + tid, s1);
}

// ---------------- FM epilogue ----------------
__global__ void fm_final(const float* __restrict__ ctile, const float* __restrict__ gbias,
                         float* __restrict__ out) {
    int b = blockIdx.x * blockDim.x + threadIdx.x;
    if (b >= B_ROWS) return;
    int t = b >> 4, m = b & 15;
    const float* t0 = ctile + (size_t)t * 512 + m * 16;
    float ss = 0.0f;
#pragma unroll
    for (int d = 0; d < 16; ++d) { float v = t0[d]; ss += v * v; }
    float sumsq = t0[256 + 0];
    float lin   = t0[256 + 1];
    out[b] = gbias[0] + lin + 0.5f * (ss - sumsq);
}

extern "C" void kernel_launch(void* const* d_in, const int* in_sizes, int n_in,
                              void* d_out, int out_size, void* d_ws, size_t ws_size,
                              hipStream_t stream) {
    const int*   x     = (const int*)d_in[0];
    const float* emb   = (const float*)d_in[1];
    const float* bias  = (const float*)d_in[2];
    const float* gbias = (const float*)d_in[3];
    float*       out   = (float*)d_out;

    _Float16* embp = (_Float16*)d_ws;                       // 5000*512 halfs = 5.12 MB
    _Float16* auxp = embp + (size_t)NCHUNKS * 512;          // 5.12 MB
    float*    ctile = (float*)((char*)d_ws + 2ull * NCHUNKS * 512 * sizeof(_Float16)); // 128 KB

    fm_zero <<<(MTILES * 512 + 255) / 256, 256, 0, stream>>>(ctile);
    fm_pack <<<(NCHUNKS * 32 + 255) / 256, 256, 0, stream>>>(emb, bias, embp, auxp);
    fm_main <<<MTILES * SLICES, 256, 0, stream>>>(x, embp, auxp, ctile);
    fm_final<<<(B_ROWS + 255) / 256, 256, 0, stream>>>(ctile, gbias, out);
}
